// VectorQuantizer_78855599555404
// MI455X (gfx1250) — matrix operand order repaired and checked by reading, not compiled, not run on hardware
//
#include <hip/hip_runtime.h>

typedef __attribute__((ext_vector_type(16))) _Float16 v16h;
typedef __attribute__((ext_vector_type(8)))  _Float16 v8h;
typedef __attribute__((ext_vector_type(8)))  float    v8f;

#define K_CODES 512
#define DIM     64
#define HW      4096                 // H*W
#define N_VEC   (64*64*64)           // B*H*W = 262144
#define ROWS_PER_WG 128
#define N_WG    (N_VEC / ROWS_PER_WG)  // 2048
#define CB_PAD  72                   // halves per codebook row in LDS (padded)
#define ZT_PAD  72                   // halves per z row in LDS (padded)
#define ZF_PAD  65                   // floats per z row in LDS (padded)

// workspace layout (bytes)
#define WS_CB16_OFF 0                // 512*64 halves = 65536 B
#define WS_E2_OFF   65536            // 512 floats   = 2048 B
#define WS_PART_OFF 67584            // 2048 floats  = 8192 B

// output layout (floats): z_q (B,D,H,W), loss, indices(B,H,W)
#define OUT_LOSS_OFF 16777216
#define OUT_IDX_OFF  (OUT_LOSS_OFF + 1)

// ---- CDNA5 async global->LDS (guarded; falls back to VGPR copy) ----------
#if defined(__has_builtin)
#if __has_builtin(__builtin_amdgcn_global_load_async_to_lds_b128)
#define VQ_HAS_ASYNC_LDS 1
#endif
#endif

#ifdef VQ_HAS_ASYNC_LDS
typedef int vq_v4i __attribute__((vector_size(16)));
typedef __attribute__((address_space(1))) vq_v4i vq_gvec;  // global (AS1)
typedef __attribute__((address_space(3))) vq_v4i vq_lvec;  // LDS (AS3)

__device__ __forceinline__ void vq_async_b128(const void* gsrc, void* ldst) {
  __builtin_amdgcn_global_load_async_to_lds_b128(
      (vq_gvec*)gsrc, (vq_lvec*)ldst, 0, 0);
}
__device__ __forceinline__ void vq_wait_async0() {
#if __has_builtin(__builtin_amdgcn_s_wait_asynccnt)
  __builtin_amdgcn_s_wait_asynccnt(0);
#else
  asm volatile("s_wait_asynccnt 0" ::: "memory");
#endif
}
#endif

// ---------------------------------------------------------------- prep ----
__global__ __launch_bounds__(256) void vq_prep_78855599(
    const float* __restrict__ cb, _Float16* __restrict__ cb16,
    float* __restrict__ e2) {
  int gid = blockIdx.x * 256 + threadIdx.x;
  if (gid < K_CODES * DIM) cb16[gid] = (_Float16)cb[gid];
  if (gid < K_CODES) {
    const float* row = cb + gid * DIM;
    float s = 0.0f;
    #pragma unroll
    for (int d = 0; d < DIM; ++d) s = fmaf(row[d], row[d], s);
    e2[gid] = s;
  }
}

// ---------------------------------------------------------------- main ----
__global__ __launch_bounds__(256) void vq_main_78855599(
    const float* __restrict__ z, const float* __restrict__ cb,
    const _Float16* __restrict__ cb16, const float* __restrict__ e2,
    float* __restrict__ zq, float* __restrict__ idx_out,
    float* __restrict__ partial) {
  __shared__ __align__(16) _Float16 sCB[K_CODES * CB_PAD];      // 73728 B
  __shared__ __align__(16) _Float16 sZ16[ROWS_PER_WG * ZT_PAD]; // 18432 B
  __shared__ __align__(16) float    sZ32[ROWS_PER_WG * ZF_PAD]; // 33280 B
  __shared__ float sE2[K_CODES];                                // 2048 B
  __shared__ int   sIdx[ROWS_PER_WG];                           // 512 B
  __shared__ float sRed[256];                                   // 1024 B

  const int tid  = threadIdx.x;
  const int lane = tid & 31;
  const int wave = tid >> 5;
  const int ln16 = lane & 15;
  const int hi16 = lane >> 4;        // 0: lanes 0-15, 1: lanes 16-31

  // ---- stage f16 codebook into padded LDS.
  // Each 16-B chunk (8 halves) is contiguous in src and in padded dst, so it
  // maps 1:1 onto per-lane async B128 copies (ASYNCcnt, no VGPR data).
#ifdef VQ_HAS_ASYNC_LDS
  for (int j = tid; j < (K_CODES * DIM) / 8; j += 256) {
    int c = j >> 3;            // code row
    int d = (j & 7) * 8;       // d offset within row
    vq_async_b128(cb16 + j * 8, &sCB[c * CB_PAD + d]);
  }
#else
  for (int j = tid; j < (K_CODES * DIM) / 8; j += 256) {
    v8h v = *reinterpret_cast<const v8h*>(cb16 + j * 8);
    int c = j >> 3;
    int d = (j & 7) * 8;
    *reinterpret_cast<v8h*>(&sCB[c * CB_PAD + d]) = v;
  }
#endif
  for (int j = tid; j < K_CODES; j += 256) sE2[j] = e2[j];

  // ---- stage z tile (128 rows x 64 dims), fp32 + f16 copies.
  // Overlaps with the codebook async DMA still in flight.
  const int n0 = blockIdx.x * ROWS_PER_WG;
  const int b  = n0 / HW;
  const int p0 = n0 % HW;
  const size_t zbase = (size_t)b * (DIM * HW) + p0;
  #pragma unroll 4
  for (int it = 0; it < (ROWS_PER_WG * DIM) / 256; ++it) {
    int e = it * 256 + tid;
    int d = e >> 7;            // 0..63
    int r = e & 127;           // 0..127
    float v = z[zbase + (size_t)d * HW + r];
    sZ32[r * ZF_PAD + d] = v;
    sZ16[r * ZT_PAD + d] = (_Float16)v;
  }
#ifdef VQ_HAS_ASYNC_LDS
  vq_wait_async0();            // this wave's codebook chunks are in LDS
#endif
  __syncthreads();

  // ---- A fragments for this wave's 16 rows (kept in VGPRs for whole sweep)
  // A 16x32 in this code's contiguous k order (same as B): lane m%16 holds row m; lanes<16: K = i,
  // lanes>=16: K = i+16.  q = dc*32 + 16*hi16, halves [q..q+15]
  const int rbase = wave * 16 + ln16;
  const int qb = hi16 * 16;
  const _Float16* zr = &sZ16[rbase * ZT_PAD];
  v8h a0lo = *reinterpret_cast<const v8h*>(zr + qb);
  v8h a0hi = *reinterpret_cast<const v8h*>(zr + qb + 8);
  v8h a1lo = *reinterpret_cast<const v8h*>(zr + 32 + qb);
  v8h a1hi = *reinterpret_cast<const v8h*>(zr + 32 + qb + 8);
  v16h a0 = __builtin_shufflevector(a0lo, a0hi, 0,1,2,3,4,5,6,7,8,9,10,11,12,13,14,15);
  v16h a1 = __builtin_shufflevector(a1lo, a1hi, 0,1,2,3,4,5,6,7,8,9,10,11,12,13,14,15);

  float bestVal[8];
  int   bestIdx[8];
  #pragma unroll
  for (int v = 0; v < 8; ++v) { bestVal[v] = 3.4e38f; bestIdx[v] = 0; }

  // ---- sweep 32 code tiles: score = ||e||^2 - 2*(z.e)  (||z||^2 irrelevant to argmin)
  const int dB = hi16 * 16;   // B layout: lanes<16 K=0..15, lanes>=16 K=16..31
  #pragma unroll 2
  for (int kt = 0; kt < K_CODES / 16; ++kt) {
    const _Float16* cr = &sCB[(kt * 16 + ln16) * CB_PAD + dB];
    v8h b0lo = *reinterpret_cast<const v8h*>(cr);
    v8h b0hi = *reinterpret_cast<const v8h*>(cr + 8);
    v8h b1lo = *reinterpret_cast<const v8h*>(cr + 32);
    v8h b1hi = *reinterpret_cast<const v8h*>(cr + 40);
    v16h b0 = __builtin_shufflevector(b0lo, b0hi, 0,1,2,3,4,5,6,7,8,9,10,11,12,13,14,15);
    v16h b1 = __builtin_shufflevector(b1lo, b1hi, 0,1,2,3,4,5,6,7,8,9,10,11,12,13,14,15);

    v8f c = {0.f,0.f,0.f,0.f,0.f,0.f,0.f,0.f};
    c = __builtin_amdgcn_wmma_f32_16x16x32_f16(false, a0, false, b0, (short)0, c, false, false);
    c = __builtin_amdgcn_wmma_f32_16x16x32_f16(false, a1, false, b1, (short)0, c, false, false);

    const float e2v  = sE2[kt * 16 + ln16];
    const int   kidx = kt * 16 + ln16;
    #pragma unroll
    for (int v = 0; v < 8; ++v) {
      float s = fmaf(-2.0f, c[v], e2v);
      bool take = s < bestVal[v];          // strict < keeps earliest (lowest) k
      bestVal[v] = take ? s : bestVal[v];
      bestIdx[v] = take ? kidx : bestIdx[v];
    }
  }

  // ---- argmin across the 16 columns (half-wave butterfly, tie -> lower idx)
  // C layout: VGPR v, lanes<16 = row v, lanes>=16 = row v+8
  #pragma unroll
  for (int v = 0; v < 8; ++v) {
    float bv = bestVal[v];
    int   bi = bestIdx[v];
    #pragma unroll
    for (int m = 8; m >= 1; m >>= 1) {
      float ov = __shfl_xor(bv, m, 32);
      int   oi = __shfl_xor(bi, m, 32);
      if (ov < bv || (ov == bv && oi < bi)) { bv = ov; bi = oi; }
    }
    if (ln16 == 0) {
      int row = wave * 16 + v + hi16 * 8;      // local row 0..127
      sIdx[row] = bi;
      idx_out[n0 + row] = (float)bi;
    }
  }
  __syncthreads();

  // ---- fused gather + z_q write + loss partial (z never re-read from HBM)
  float acc = 0.0f;
  #pragma unroll 4
  for (int it = 0; it < (ROWS_PER_WG * DIM) / 256; ++it) {
    int e = it * 256 + tid;
    int d = e >> 7;
    int r = e & 127;
    int k = sIdx[r];
    float cv = cb[k * DIM + d];          // fp32 gather, codebook is L2-resident
    float ze = sZ32[r * ZF_PAD + d];
    float diff = cv - ze;
    zq[zbase + (size_t)d * HW + r] = ze + diff;   // mimic z_e + (z_q - z_e)
    acc = fmaf(diff, diff, acc);
  }
  sRed[tid] = acc;
  __syncthreads();
  #pragma unroll
  for (int s = 128; s > 0; s >>= 1) {
    if (tid < s) sRed[tid] += sRed[tid + s];
    __syncthreads();
  }
  if (tid == 0) partial[blockIdx.x] = sRed[0];
}

// ------------------------------------------------------------- finalize ----
__global__ __launch_bounds__(256) void vq_finalize_78855599(
    const float* __restrict__ partial, float* __restrict__ loss) {
  __shared__ float red[256];
  int tid = threadIdx.x;
  float a = 0.0f;
  for (int i = tid; i < N_WG; i += 256) a += partial[i];
  red[tid] = a;
  __syncthreads();
  #pragma unroll
  for (int s = 128; s > 0; s >>= 1) {
    if (tid < s) red[tid] += red[tid + s];
    __syncthreads();
  }
  if (tid == 0) loss[0] = 1.25f * red[0] * (1.0f / 16777216.0f);
}

// ---------------------------------------------------------------- launch ----
extern "C" void kernel_launch(void* const* d_in, const int* in_sizes, int n_in,
                              void* d_out, int out_size, void* d_ws, size_t ws_size,
                              hipStream_t stream) {
  const float* z  = (const float*)d_in[0];   // (64,64,64,64) fp32
  const float* cb = (const float*)d_in[1];   // (512,64) fp32

  float* out   = (float*)d_out;
  float* zq    = out;
  float* lossp = out + OUT_LOSS_OFF;
  float* idxp  = out + OUT_IDX_OFF;

  _Float16* cb16 = (_Float16*)((char*)d_ws + WS_CB16_OFF);
  float* e2      = (float*)((char*)d_ws + WS_E2_OFF);
  float* partial = (float*)((char*)d_ws + WS_PART_OFF);

  vq_prep_78855599<<<(K_CODES * DIM + 255) / 256, 256, 0, stream>>>(cb, cb16, e2);
  vq_main_78855599<<<N_WG, 256, 0, stream>>>(z, cb, cb16, e2, zq, idxp, partial);
  vq_finalize_78855599<<<1, 256, 0, stream>>>(partial, lossp);
}
